// MechanismTransformerBlock_56899726737903
// MI455X (gfx1250) — matrix operand order repaired and checked by reading, not compiled, not run on hardware
//
#include <hip/hip_runtime.h>
#include <hip/hip_bf16.h>

typedef __attribute__((ext_vector_type(16))) __bf16 v16bf;
typedef __attribute__((ext_vector_type(8)))  float  v8f;
typedef __attribute__((ext_vector_type(4)))  int    vi4;

struct alignas(16) B16x8 { unsigned int u[4]; };   // 8 bf16 / 16 bytes

#if defined(__has_builtin)
#  if __has_builtin(__builtin_amdgcn_global_load_async_to_lds_b128)
#    define USE_ASYNC_COPY 1
#  endif
#endif
#ifndef USE_ASYNC_COPY
#  define USE_ASYNC_COPY 0
#endif

static __device__ __forceinline__ void async_cp16(const void* g, void* l){
#if USE_ASYNC_COPY
  __builtin_amdgcn_global_load_async_to_lds_b128(
      (vi4 __attribute__((address_space(1))) *)(g),
      (vi4 __attribute__((address_space(3))) *)(l), 0, 0);
#else
  (void)g; (void)l;
#endif
}

static __device__ __forceinline__ void wait_async0(){
#if USE_ASYNC_COPY
#  if __has_builtin(__builtin_amdgcn_s_wait_asynccnt)
  __builtin_amdgcn_s_wait_asynccnt(0);
#  else
  asm volatile("s_wait_asynccnt 0x0" ::: "memory");
#  endif
#endif
}

static __device__ __forceinline__ float gelu_f(float v){
  return 0.5f * v * (1.0f + erff(v * 0.70710678118654752f));
}

// ---------------------------------------------------------------- utilities
__global__ void cvt_f32_bf16(const float* __restrict__ s, __bf16* __restrict__ d, int n){
  int i = (blockIdx.x * blockDim.x + threadIdx.x) * 4;
  #pragma unroll
  for (int t = 0; t < 4; ++t) if (i + t < n) d[i + t] = (__bf16)s[i + t];
}

__global__ void softmax_ew_kernel(const float* __restrict__ ew,
                                  float* __restrict__ wout, float* __restrict__ wsm){
  if (threadIdx.x == 0){
    float m = ew[0];
    for (int e = 1; e < 5; ++e) m = fmaxf(m, ew[e]);
    float v[5], s = 0.f;
    for (int e = 0; e < 5; ++e){ v[e] = expf(ew[e] - m); s += v[e]; }
    for (int e = 0; e < 5; ++e){ float r = v[e] / s; wout[e] = r; wsm[e] = r; }
  }
}

__global__ void combine_w5(const float* __restrict__ W5, const float* __restrict__ wsm,
                           __bf16* __restrict__ Wc, int n){
  int i = (blockIdx.x * blockDim.x + threadIdx.x) * 4;
  float w0 = wsm[0], w1 = wsm[1], w2 = wsm[2], w3 = wsm[3], w4 = wsm[4];
  #pragma unroll
  for (int t = 0; t < 4; ++t){
    int j = i + t;
    if (j < n){
      float s = w0*W5[j] + w1*W5[(size_t)n + j] + w2*W5[2*(size_t)n + j]
              + w3*W5[3*(size_t)n + j] + w4*W5[4*(size_t)n + j];
      Wc[j] = (__bf16)s;
    }
  }
}

__global__ void combine_b5(const float* __restrict__ b5, const float* __restrict__ wsm,
                           float* __restrict__ bc){
  int i = blockIdx.x * blockDim.x + threadIdx.x;
  if (i < 1024){
    float s = 0.f;
    for (int e = 0; e < 5; ++e) s += wsm[e] * b5[e * 1024 + i];
    bc[i] = s;
  }
}

__global__ void build_fused(const float* __restrict__ pos, const float* __restrict__ tim,
                            __bf16* __restrict__ fused){
  int i = blockIdx.x * blockDim.x + threadIdx.x;     // < 1M
  int t = i >> 10, c = i & 1023;
  float v = (c < 512) ? pos[t * 512 + c] : tim[t * 512 + (c - 512)];
  fused[i] = (__bf16)v;
}

__global__ void add_embed(const float* __restrict__ x, const float* __restrict__ fp,
                          float* __restrict__ x1){
  int i = blockIdx.x * blockDim.x + threadIdx.x;     // 4M
  x1[i] = x[i] + fp[i & 1048575];
}

// ------------------------------------------------------------ LayerNorm -> bf16
__global__ __launch_bounds__(256) void layernorm_to_bf16(
    const float* __restrict__ x, const float* __restrict__ g,
    const float* __restrict__ b, __bf16* __restrict__ out){
  const int C = 1024;
  int row = blockIdx.x;
  const float* xr = x + (size_t)row * C;
  float v[4], s1 = 0.f, s2 = 0.f;
  #pragma unroll
  for (int i = 0; i < 4; ++i){ v[i] = xr[threadIdx.x + 256*i]; s1 += v[i]; s2 += v[i]*v[i]; }
  #pragma unroll
  for (int o = 16; o > 0; o >>= 1){ s1 += __shfl_xor(s1, o, 32); s2 += __shfl_xor(s2, o, 32); }
  __shared__ float red[16];
  int wid = threadIdx.x >> 5;
  if ((threadIdx.x & 31) == 0){ red[wid] = s1; red[8 + wid] = s2; }
  __syncthreads();
  float t1 = 0.f, t2 = 0.f;
  #pragma unroll
  for (int q = 0; q < 8; ++q){ t1 += red[q]; t2 += red[8 + q]; }
  float mean = t1 * (1.f / C);
  float var  = t2 * (1.f / C) - mean * mean;
  float rstd = rsqrtf(var + 1e-5f);
  __bf16* orow = out + (size_t)row * C;
  #pragma unroll
  for (int i = 0; i < 4; ++i){
    int c = threadIdx.x + 256*i;
    orow[c] = (__bf16)((v[i] - mean) * rstd * g[c] + b[c]);
  }
}

// ------------------------------------------------------- generic bf16 WMMA GEMM
// out[M,N] = act(A[M,K] @ W[N,K]^T + bias) + resid ; A,W bf16, acc fp32
// 64x128 block tile, BK=32, 8 waves x (16x64), double-buffered async staging.
__global__ __launch_bounds__(256) void gemm_bf16_wmma(
    const __bf16* __restrict__ Amat, const __bf16* __restrict__ Wmat,
    const float* __restrict__ bias, const float* __restrict__ resid,
    float* __restrict__ outF, __bf16* __restrict__ outB,
    int M, int N, int K, int act)
{
#if USE_ASYNC_COPY
  __shared__ __bf16 As[2][64][32];
  __shared__ __bf16 Bs[2][128][32];
#else
  __shared__ __bf16 As[1][64][32];
  __shared__ __bf16 Bs[1][128][32];
#endif
  const int tid = threadIdx.x;
  const int lane = tid & 31, wv = tid >> 5;
  const int hi = lane >> 4, l15 = lane & 15;
  const int m0 = blockIdx.y * 64;
  const int n0 = blockIdx.x * 128;

  const int arow_l = tid >> 2;        // 0..63
  const int acol_l = (tid & 3) * 8;   // 0,8,16,24
  const __bf16* aptr  = Amat + (size_t)(m0 + arow_l) * K + acol_l;
  const __bf16* bptr0 = Wmat + (size_t)(n0 + arow_l) * K + acol_l;
  const __bf16* bptr1 = Wmat + (size_t)(n0 + 64 + arow_l) * K + acol_l;

  const v8f zero8 = {0.f,0.f,0.f,0.f,0.f,0.f,0.f,0.f};
  v8f acc[4]; acc[0]=zero8; acc[1]=zero8; acc[2]=zero8; acc[3]=zero8;

  const int nkt = K >> 5;

#if USE_ASYNC_COPY
  // prologue: async-stage tile 0 straight into LDS (no VGPR round trip)
  async_cp16(aptr,  &As[0][arow_l][acol_l]);
  async_cp16(bptr0, &Bs[0][arow_l][acol_l]);
  async_cp16(bptr1, &Bs[0][64 + arow_l][acol_l]);
#else
  B16x8 ra  = *(const B16x8*)aptr;
  B16x8 rb0 = *(const B16x8*)bptr0;
  B16x8 rb1 = *(const B16x8*)bptr1;
#endif

  for (int kt = 0; kt < nkt; ++kt){
#if USE_ASYNC_COPY
    const int cur = kt & 1;
    wait_async0();          // my async tile writes landed
    __syncthreads();        // everyone's landed; prev reads of other buf done
    if (kt + 1 < nkt){
      const int nxt = cur ^ 1;
      async_cp16(aptr  + (size_t)(kt + 1) * 32, &As[nxt][arow_l][acol_l]);
      async_cp16(bptr0 + (size_t)(kt + 1) * 32, &Bs[nxt][arow_l][acol_l]);
      async_cp16(bptr1 + (size_t)(kt + 1) * 32, &Bs[nxt][64 + arow_l][acol_l]);
    }
#else
    const int cur = 0;
    *(B16x8*)(&As[0][arow_l][acol_l])      = ra;
    *(B16x8*)(&Bs[0][arow_l][acol_l])      = rb0;
    *(B16x8*)(&Bs[0][64 + arow_l][acol_l]) = rb1;
    __syncthreads();
    if (kt + 1 < nkt){   // issue next tile's loads before compute (overlap)
      ra  = *(const B16x8*)(aptr  + (size_t)(kt + 1) * 32);
      rb0 = *(const B16x8*)(bptr0 + (size_t)(kt + 1) * 32);
      rb1 = *(const B16x8*)(bptr1 + (size_t)(kt + 1) * 32);
      __builtin_prefetch(aptr  + (size_t)(kt + 2) * 32, 0, 3);
      __builtin_prefetch(bptr0 + (size_t)(kt + 2) * 32, 0, 3);
      __builtin_prefetch(bptr1 + (size_t)(kt + 2) * 32, 0, 3);
    }
#endif

    // A fragment 16x32 (contiguous K: lane half h holds k = 16h..16h+15)
    union { v16bf v; unsigned int u[8]; } af;
    const __bf16* arow = &As[cur][(wv & 3) * 16 + l15][0];
    #pragma unroll
    for (int i = 0; i < 8; ++i){
      int kb = 2*i + hi * 16;
      af.u[i] = *(const unsigned int*)(arow + kb);
    }
    #pragma unroll
    for (int j = 0; j < 4; ++j){
      const __bf16* brow = &Bs[cur][(wv >> 2) * 64 + j * 16 + l15][hi * 16];
      v16bf bf = *(const v16bf*)brow;
      acc[j] = __builtin_amdgcn_wmma_f32_16x16x32_bf16(false, af.v, false, bf,
                                                       (short)0, acc[j], false, false);
    }
#if !USE_ASYNC_COPY
    __syncthreads();
#endif
  }

  #pragma unroll
  for (int j = 0; j < 4; ++j){
    int col = n0 + (wv >> 2) * 64 + j * 16 + l15;
    float bv = bias ? bias[col] : 0.f;
    #pragma unroll
    for (int r = 0; r < 8; ++r){
      int row = m0 + (wv & 3) * 16 + hi * 8 + r;
      float v = acc[j][r] + bv;
      if (act == 1) v = gelu_f(v);
      size_t idx = (size_t)row * N + col;
      if (resid) v += resid[idx];
      if (outF) outF[idx] = v;
      if (outB) outB[idx] = (__bf16)v;
    }
  }
}

// ------------------------------------------------------------- mechanism gate
__global__ __launch_bounds__(256) void mech_kernel(
    const __bf16* __restrict__ m1, const float* __restrict__ Wm2,
    const float* __restrict__ bm2, float* __restrict__ mech_ws,
    float* __restrict__ mech_out){
  int row  = blockIdx.x * 8 + (threadIdx.x >> 5);
  int lane = threadIdx.x & 31;
  const __bf16* mr = m1 + (size_t)row * 512;
  float s = 0.f;
  #pragma unroll
  for (int i = 0; i < 16; ++i){
    int c = lane + 32 * i;
    s += (float)mr[c] * Wm2[c];
  }
  #pragma unroll
  for (int o = 16; o > 0; o >>= 1) s += __shfl_xor(s, o, 32);
  if (lane == 0){
    float t = 1.f / (1.f + expf(-(s + bm2[0])));
    mech_ws[row] = t; mech_out[row] = t;
  }
}

// ----------------------------------------------------------- flash attention
__global__ __launch_bounds__(256) void attention_kernel(
    const __bf16* __restrict__ qg, const __bf16* __restrict__ kg,
    const __bf16* __restrict__ vg, const float* __restrict__ mech,
    __bf16* __restrict__ ctx)
{
  const int T = 1024, C = 1024;
  __shared__ __bf16 Ks[32][64];
  __shared__ __bf16 Vt[64][32];
  __shared__ __bf16 Ps[8][16][32];
  __shared__ float  msc[32];

  const int tid = threadIdx.x, lane = tid & 31, wv = tid >> 5;
  const int hi = lane >> 4, l15 = lane & 15;
  const int qblk = blockIdx.x & 7;
  const int bh   = blockIdx.x >> 3;
  const int b = bh >> 4, hd = bh & 15;
  const int qt0 = qblk * 128 + wv * 16;
  const size_t headoff = (size_t)hd * 64;

  // Q fragments: 16x64 -> two 16x32 A-frags, kept in registers
  union { v16bf v; unsigned int u[8]; } qa0, qa1;
  {
    const __bf16* qrow = qg + ((size_t)(b * T + qt0 + l15)) * C + headoff;
    #pragma unroll
    for (int i = 0; i < 8; ++i){
      int kb = 2*i + hi * 16;
      qa0.u[i] = *(const unsigned int*)(qrow + kb);
      qa1.u[i] = *(const unsigned int*)(qrow + 32 + kb);
    }
  }

  const v8f zero8 = {0.f,0.f,0.f,0.f,0.f,0.f,0.f,0.f};
  v8f o0 = zero8, o1 = zero8, o2 = zero8, o3 = zero8;
  float mrow[8], lsum[8];
  #pragma unroll
  for (int r = 0; r < 8; ++r){ mrow[r] = -1e30f; lsum[r] = 0.f; }

  const int srow = tid >> 3;        // 0..31 (key row in tile)
  const int c8   = (tid & 7) * 8;   // 0..56 (d chunk)

  for (int kt = 0; kt < 32; ++kt){
    __syncthreads();
    const size_t sbase = ((size_t)(b * T + kt * 32 + srow)) * C + headoff;
    // issue all global loads first, then store to LDS (single wait)
    B16x8 kreg = *(const B16x8*)(kg + sbase + c8);
    union { B16x8 q; __bf16 e[8]; } vv;
    vv.q = *(const B16x8*)(vg + sbase + c8);
    float mval = (tid < 32) ? mech[b * T + kt * 32 + tid] : 0.f;
    *(B16x8*)(&Ks[srow][c8]) = kreg;
    #pragma unroll
    for (int q = 0; q < 8; ++q) Vt[c8 + q][srow] = vv.e[q];   // transpose V
    if (tid < 32) msc[tid] = 1.f + mval;
    __syncthreads();

    // S = Q K^T : 16 queries x 32 keys as two 16x16 accumulators
    v8f s0 = zero8, s1 = zero8;
    {
      const __bf16* k0 = &Ks[l15][hi * 16];
      v16bf b0lo = *(const v16bf*)(k0);
      v16bf b0hi = *(const v16bf*)(k0 + 32);
      s0 = __builtin_amdgcn_wmma_f32_16x16x32_bf16(false, qa0.v, false, b0lo, (short)0, s0, false, false);
      s0 = __builtin_amdgcn_wmma_f32_16x16x32_bf16(false, qa1.v, false, b0hi, (short)0, s0, false, false);
      const __bf16* k1 = &Ks[16 + l15][hi * 16];
      v16bf b1lo = *(const v16bf*)(k1);
      v16bf b1hi = *(const v16bf*)(k1 + 32);
      s1 = __builtin_amdgcn_wmma_f32_16x16x32_bf16(false, qa0.v, false, b1lo, (short)0, s1, false, false);
      s1 = __builtin_amdgcn_wmma_f32_16x16x32_bf16(false, qa1.v, false, b1hi, (short)0, s1, false, false);
    }
    float f0 = 0.125f * msc[l15];
    float f1 = 0.125f * msc[16 + l15];
    float alpha[8];
    #pragma unroll
    for (int r = 0; r < 8; ++r){
      float a = s0[r] * f0, c = s1[r] * f1;
      float mx = fmaxf(a, c);
      #pragma unroll
      for (int o = 8; o > 0; o >>= 1) mx = fmaxf(mx, __shfl_xor(mx, o, 32));
      float mnew = fmaxf(mrow[r], mx);
      float al = expf(mrow[r] - mnew);
      float p0 = expf(a - mnew), p1 = expf(c - mnew);
      float rs = p0 + p1;
      #pragma unroll
      for (int o = 8; o > 0; o >>= 1) rs += __shfl_xor(rs, o, 32);
      lsum[r] = lsum[r] * al + rs;
      mrow[r] = mnew; alpha[r] = al;
      s0[r] = p0; s1[r] = p1;
    }
    #pragma unroll
    for (int r = 0; r < 8; ++r){ o0[r]*=alpha[r]; o1[r]*=alpha[r]; o2[r]*=alpha[r]; o3[r]*=alpha[r]; }

    // P -> LDS (bf16), re-fragment as A matrix
    #pragma unroll
    for (int r = 0; r < 8; ++r){
      Ps[wv][r + hi * 8][l15]      = (__bf16)s0[r];
      Ps[wv][r + hi * 8][16 + l15] = (__bf16)s1[r];
    }
    __syncthreads();
    union { v16bf v; unsigned int u[8]; } pa;
    #pragma unroll
    for (int i = 0; i < 8; ++i){
      int kb = 2*i + hi * 16;
      pa.u[i] = *(const unsigned int*)(&Ps[wv][l15][kb]);
    }
    {
      v16bf vb0 = *(const v16bf*)(&Vt[ 0 + l15][hi * 16]);
      v16bf vb1 = *(const v16bf*)(&Vt[16 + l15][hi * 16]);
      v16bf vb2 = *(const v16bf*)(&Vt[32 + l15][hi * 16]);
      v16bf vb3 = *(const v16bf*)(&Vt[48 + l15][hi * 16]);
      o0 = __builtin_amdgcn_wmma_f32_16x16x32_bf16(false, pa.v, false, vb0, (short)0, o0, false, false);
      o1 = __builtin_amdgcn_wmma_f32_16x16x32_bf16(false, pa.v, false, vb1, (short)0, o1, false, false);
      o2 = __builtin_amdgcn_wmma_f32_16x16x32_bf16(false, pa.v, false, vb2, (short)0, o2, false, false);
      o3 = __builtin_amdgcn_wmma_f32_16x16x32_bf16(false, pa.v, false, vb3, (short)0, o3, false, false);
    }
  }

  #pragma unroll
  for (int r = 0; r < 8; ++r){
    float inv = 1.f / lsum[r];
    size_t rowoff = ((size_t)(b * T + qt0 + r + hi * 8)) * C + headoff;
    ctx[rowoff +  0 + l15] = (__bf16)(o0[r] * inv);
    ctx[rowoff + 16 + l15] = (__bf16)(o1[r] * inv);
    ctx[rowoff + 32 + l15] = (__bf16)(o2[r] * inv);
    ctx[rowoff + 48 + l15] = (__bf16)(o3[r] * inv);
  }
}

// ------------------------------------------------------------------ launcher
extern "C" void kernel_launch(void* const* d_in, const int* in_sizes, int n_in,
                              void* d_out, int out_size, void* d_ws, size_t ws_size,
                              hipStream_t stream) {
  (void)in_sizes; (void)n_in; (void)out_size; (void)ws_size;
  const int B = 4, T = 1024, C = 1024, BT = B * T, DFF = 4096;

  const float* x    = (const float*)d_in[0];
  const float* pos  = (const float*)d_in[1];
  const float* tim  = (const float*)d_in[2];
  const float* Wint = (const float*)d_in[3];
  const float* bint = (const float*)d_in[4];
  const float* g1   = (const float*)d_in[5];
  const float* be1  = (const float*)d_in[6];
  const float* Wq   = (const float*)d_in[7];
  const float* bq   = (const float*)d_in[8];
  const float* Wk   = (const float*)d_in[9];
  const float* bk   = (const float*)d_in[10];
  const float* Wv   = (const float*)d_in[11];
  const float* bv   = (const float*)d_in[12];
  const float* Wo   = (const float*)d_in[13];
  const float* bo   = (const float*)d_in[14];
  const float* Wm1  = (const float*)d_in[15];
  const float* bm1  = (const float*)d_in[16];
  const float* Wm2  = (const float*)d_in[17];
  const float* bm2  = (const float*)d_in[18];
  const float* g2   = (const float*)d_in[19];
  const float* be2  = (const float*)d_in[20];
  const float* W5   = (const float*)d_in[21];
  const float* b5   = (const float*)d_in[22];
  const float* ew   = (const float*)d_in[23];
  const float* g3   = (const float*)d_in[24];
  const float* be3  = (const float*)d_in[25];
  const float* Wf1  = (const float*)d_in[26];
  const float* bf1  = (const float*)d_in[27];
  const float* Wf2  = (const float*)d_in[28];
  const float* bf2  = (const float*)d_in[29];

  float* out      = (float*)d_out;
  float* out_mech = out + (size_t)BT * C;       // 4,194,304
  float* out_w    = out_mech + BT;              // +4096

  size_t off = 0;
  auto alloc = [&](size_t bytes) -> void* {
    void* p = (void*)((char*)d_ws + off);
    off += (bytes + 255) & ~(size_t)255;
    return p;
  };
  float*  wsm    = (float*) alloc(64);
  float*  mechv  = (float*) alloc((size_t)BT * 4);
  __bf16* fused  = (__bf16*)alloc((size_t)T * C * 2);
  float*  fproj  = (float*) alloc((size_t)T * C * 4);
  float*  x1     = (float*) alloc((size_t)BT * C * 4);
  __bf16* hb     = (__bf16*)alloc((size_t)BT * C * 2);
  __bf16* qb     = (__bf16*)alloc((size_t)BT * C * 2);   // qb..ctxb contiguous (32MB)
  __bf16* kb     = (__bf16*)alloc((size_t)BT * C * 2);
  __bf16* vb     = (__bf16*)alloc((size_t)BT * C * 2);
  __bf16* ctxb   = (__bf16*)alloc((size_t)BT * C * 2);
  __bf16* m1     = (__bf16*)alloc((size_t)BT * 512 * 2);
  __bf16* WintB  = (__bf16*)alloc((size_t)C * C * 2);
  __bf16* WqB    = (__bf16*)alloc((size_t)C * C * 2);
  __bf16* WkB    = (__bf16*)alloc((size_t)C * C * 2);
  __bf16* WvB    = (__bf16*)alloc((size_t)C * C * 2);
  __bf16* WoB    = (__bf16*)alloc((size_t)C * C * 2);
  __bf16* Wm1B   = (__bf16*)alloc((size_t)512 * C * 2);
  __bf16* Wf1B   = (__bf16*)alloc((size_t)DFF * C * 2);
  __bf16* Wf2B   = (__bf16*)alloc((size_t)C * DFF * 2);
  __bf16* WcB    = (__bf16*)alloc((size_t)C * C * 2);
  float*  bc     = (float*) alloc((size_t)C * 4);
  __bf16* f1     = qb;    // reuse dead q/k/v/ctx region (32MB) for FFN hidden

  const int CC = C * C;   // 1,048,576
  // weight converts
  cvt_f32_bf16<<<CC/1024, 256, 0, stream>>>(Wint, WintB, CC);
  cvt_f32_bf16<<<CC/1024, 256, 0, stream>>>(Wq,   WqB,   CC);
  cvt_f32_bf16<<<CC/1024, 256, 0, stream>>>(Wk,   WkB,   CC);
  cvt_f32_bf16<<<CC/1024, 256, 0, stream>>>(Wv,   WvB,   CC);
  cvt_f32_bf16<<<CC/1024, 256, 0, stream>>>(Wo,   WoB,   CC);
  cvt_f32_bf16<<<(512*C)/1024, 256, 0, stream>>>(Wm1, Wm1B, 512*C);
  cvt_f32_bf16<<<(DFF*C)/1024, 256, 0, stream>>>(Wf1, Wf1B, DFF*C);
  cvt_f32_bf16<<<(DFF*C)/1024, 256, 0, stream>>>(Wf2, Wf2B, DFF*C);

  // five-element weight combine (w = softmax(ew) is also output #3)
  softmax_ew_kernel<<<1, 32, 0, stream>>>(ew, out_w, wsm);
  combine_w5<<<CC/1024, 256, 0, stream>>>(W5, wsm, WcB, CC);
  combine_b5<<<4, 256, 0, stream>>>(b5, wsm, bc);

  // x1 = x + concat(pos,time) @ Wint^T + bint
  build_fused<<<(T*C)/256, 256, 0, stream>>>(pos, tim, fused);
  gemm_bf16_wmma<<<dim3(C/128, T/64), 256, 0, stream>>>(
      fused, WintB, bint, nullptr, fproj, nullptr, T, C, C, 0);
  add_embed<<<(BT*C)/256, 256, 0, stream>>>(x, fproj, x1);

  // attention branch
  layernorm_to_bf16<<<BT, 256, 0, stream>>>(x1, g1, be1, hb);
  gemm_bf16_wmma<<<dim3(C/128, BT/64), 256, 0, stream>>>(hb, WqB, bq, nullptr, nullptr, qb, BT, C, C, 0);
  gemm_bf16_wmma<<<dim3(C/128, BT/64), 256, 0, stream>>>(hb, WkB, bk, nullptr, nullptr, kb, BT, C, C, 0);
  gemm_bf16_wmma<<<dim3(C/128, BT/64), 256, 0, stream>>>(hb, WvB, bv, nullptr, nullptr, vb, BT, C, C, 0);
  gemm_bf16_wmma<<<dim3(512/128, BT/64), 256, 0, stream>>>(hb, Wm1B, bm1, nullptr, nullptr, m1, BT, 512, C, 1);
  mech_kernel<<<BT/8, 256, 0, stream>>>(m1, Wm2, bm2, mechv, out_mech);
  attention_kernel<<<B*16*(T/128), 256, 0, stream>>>(qb, kb, vb, mechv, ctxb);
  gemm_bf16_wmma<<<dim3(C/128, BT/64), 256, 0, stream>>>(ctxb, WoB, bo, x1, x1, nullptr, BT, C, C, 0);

  // five-elements (collapsed to one GEMM)
  layernorm_to_bf16<<<BT, 256, 0, stream>>>(x1, g2, be2, hb);
  gemm_bf16_wmma<<<dim3(C/128, BT/64), 256, 0, stream>>>(hb, WcB, bc, x1, x1, nullptr, BT, C, C, 0);

  // FFN
  layernorm_to_bf16<<<BT, 256, 0, stream>>>(x1, g3, be3, hb);
  gemm_bf16_wmma<<<dim3(DFF/128, BT/64), 256, 0, stream>>>(hb, Wf1B, bf1, nullptr, nullptr, f1, BT, DFF, C, 1);
  gemm_bf16_wmma<<<dim3(C/128, BT/64), 256, 0, stream>>>(f1, Wf2B, bf2, x1, out, nullptr, BT, C, DFF, 0);
}
